// LegalHeteroGNN_7687991460339
// MI455X (gfx1250) — matrix operand order repaired and checked by reading, not compiled, not run on hardware
//
#include <hip/hip_runtime.h>
#include <stdint.h>

#define HID 128
#define HEADS 4
#define CH 32
#define BK 128   // K-chunk staged in LDS; double-buffered: 2 x 128 cols x 128 k x bf16 = 64 KB

typedef __attribute__((ext_vector_type(16))) __bf16 v16bf;
typedef __attribute__((ext_vector_type(8)))  float  v8f;

// ---------------------------------------------------------------------------
// utility kernels
// ---------------------------------------------------------------------------
__global__ void zero_f4_kernel(float4* p, long n4) {
  long i = (long)blockIdx.x * blockDim.x + threadIdx.x;
  if (i < n4) p[i] = make_float4(0.f, 0.f, 0.f, 0.f);
}

// Wt[n*K + k] = bf16(W[k*N + n])  (transpose to column-major bf16)
__global__ void prep_wt_kernel(const float* __restrict__ W, __bf16* __restrict__ Wt,
                               int K, int N) {
  int i = blockIdx.x * blockDim.x + threadIdx.x;
  if (i >= K * N) return;
  int k = i / N, n = i % N;
  Wt[(size_t)n * K + k] = (__bf16)W[i];
}

// ---------------------------------------------------------------------------
// bf16 WMMA GEMM: C[M x 128] = A[M x K](f32->bf16) * W[K x 128]  (+bias)(relu)(+=)
// Wt is pre-transposed column-major bf16 [128][K].  K % 128 == 0.
// Block = 256 threads = 8 waves; wave computes a 16x128 tile, 8 WMMAs per K-step.
// Weight chunks stream through a double-buffered LDS stage via gfx1250 async
// global->LDS DMA (global_load_async_to_lds_b128 / ASYNCcnt): chunk c+1's DMA
// overlaps chunk c's WMMAs. B-fragments are software-pipelined out of LDS so
// each WMMA only waits on <=1 outstanding DS load.
// flags: bit0 = relu epilogue, bit1 = accumulate into C.
// ---------------------------------------------------------------------------
__global__ __launch_bounds__(256)
void gemm_bf16_wmma_kernel(const float* __restrict__ A, const __bf16* __restrict__ Wt,
                           const float* __restrict__ bias, float* __restrict__ C,
                           int M, int K, int flags) {
  extern __shared__ __bf16 Bs[];           // dynamic LDS -> base offset 0; 2 buffers
  const int tid  = threadIdx.x;
  const int lane = tid & 31;
  const int wave = tid >> 5;
  const int row0 = (blockIdx.x * 8 + wave) * 16;
  const bool active = row0 < M;            // inactive waves still run staging/barriers
  const int ml = lane & 15;
  const int hi = lane >> 4;
  const int arow = active ? ((row0 + ml < M) ? (row0 + ml) : (M - 1)) : 0;
  const float* Abase = A + (size_t)arow * K;

  // staging map: thread copies 128 B of one weight column half:
  // col = tid>>1 (0..127), half = tid&1 -> Wt[col*K + k0 + half*64 .. +64)
  const int scol  = tid >> 1;
  const int shalf = tid & 1;
  const unsigned ldsCopyOff = (unsigned)((scol * BK + shalf * 64) * 2);  // bytes

  auto issue_chunk = [&](int buf, int k0) {
    const __bf16* gsrc = Wt + (size_t)scol * K + k0 + shalf * 64;
    const unsigned ldst = ldsCopyOff + (unsigned)buf * (BK * HID * 2);
#pragma unroll
    for (int i = 0; i < 8; ++i) {
      // INST_OFFSET applies to both LDS and global addresses (ISA async ops)
      asm volatile("global_load_async_to_lds_b128 %0, %1, off offset:%2"
                   :: "v"(ldst), "v"(gsrc), "i"(i * 16) : "memory");
    }
  };

  v8f acc[8] = {};
  issue_chunk(0, 0);                       // prologue: DMA chunk 0 into buffer 0
  const int nch = K / BK;
  for (int c = 0; c < nch; ++c) {
    asm volatile("s_wait_asynccnt 0x0" ::: "memory");  // own chunk-c transfers landed
    __syncthreads();                                   // all waves' chunk c visible;
                                                       // also: everyone done reading
                                                       // the other buffer (chunk c-1)
    if (c + 1 < nch) issue_chunk((c + 1) & 1, (c + 1) * BK);  // overlap DMA with WMMAs

    if (active) {
      const __bf16* Bb = Bs + (size_t)(c & 1) * (BK * HID);
      const float* Ab = Abase + c * BK;
#pragma unroll
      for (int kk0 = 0; kk0 < BK; kk0 += 32) {
        // A fragment (contiguous k order, not ISA table order): elems 0-15 <- K=16*hi+0..15
        // (matches B). Four independent b128 loads, then convert.
        const float4 q0 = *(const float4*)(Ab + kk0 + hi * 16 + 0);
        const float4 q1 = *(const float4*)(Ab + kk0 + hi * 16 + 4);
        const float4 q2 = *(const float4*)(Ab + kk0 + hi * 16 + 8);
        const float4 q3 = *(const float4*)(Ab + kk0 + hi * 16 + 12);
        v16bf a;
        a[0]  = (__bf16)q0.x; a[1]  = (__bf16)q0.y; a[2]  = (__bf16)q0.z; a[3]  = (__bf16)q0.w;
        a[4]  = (__bf16)q1.x; a[5]  = (__bf16)q1.y; a[6]  = (__bf16)q1.z; a[7]  = (__bf16)q1.w;
        a[8]  = (__bf16)q2.x; a[9]  = (__bf16)q2.y; a[10] = (__bf16)q2.z; a[11] = (__bf16)q2.w;
        a[12] = (__bf16)q3.x; a[13] = (__bf16)q3.y; a[14] = (__bf16)q3.z; a[15] = (__bf16)q3.w;

        // B fragments (32x16 bf16): lane = column n = t*16+ml, K = 16*hi..+15 ->
        // one contiguous 32-byte LDS run; pipelined one tile ahead of the WMMA.
        const __bf16* bptr = Bb + (size_t)ml * BK + kk0 + hi * 16;
        v16bf bcur = *(const v16bf*)(bptr);
#pragma unroll
        for (int t = 0; t < 8; ++t) {
          v16bf bnext = bcur;
          if (t < 7) bnext = *(const v16bf*)(bptr + (size_t)(t + 1) * 16 * BK);
          acc[t] = __builtin_amdgcn_wmma_f32_16x16x32_bf16(
              false, a, false, bcur, (short)0, acc[t], false, false);
          bcur = bnext;
        }
      }
    }
  }
  if (!active) return;

  // D layout: VGPR v -> M = row0 + v + 8*hi, N = t*16 + ml
#pragma unroll
  for (int t = 0; t < 8; ++t) {
    const int n = t * 16 + ml;
    const float bv = bias ? bias[n] : 0.f;
#pragma unroll
    for (int v = 0; v < 8; ++v) {
      const int m = row0 + v + 8 * hi;
      if (m < M) {
        float val = acc[t][v] + bv;
        if (flags & 1) val = fmaxf(val, 0.f);
        float* dst = C + (size_t)m * HID + n;
        if (flags & 2) *dst += val; else *dst = val;
      }
    }
  }
}

// a[n][h] = sum_c h[n][h*32+c] * att[h][c]
__global__ void att_reduce_kernel(const float* __restrict__ h, const float* __restrict__ att,
                                  float* __restrict__ a, int Nn) {
  int i = blockIdx.x * blockDim.x + threadIdx.x;
  if (i >= Nn * HEADS) return;
  int n = i >> 2, hh = i & 3;
  const float* hp = h + (size_t)n * HID + hh * CH;
  const float* ap = att + hh * CH;
  float s = 0.f;
#pragma unroll
  for (int c = 0; c < CH; ++c) s += hp[c] * ap[c];
  a[i] = s;
}

// order-preserving float<->u32 key for atomicMax
__device__ __forceinline__ unsigned f2key(float f) {
  unsigned u = __float_as_uint(f);
  return (u & 0x80000000u) ? ~u : (u | 0x80000000u);
}
__device__ __forceinline__ float key2f(unsigned k) {
  unsigned u = (k & 0x80000000u) ? (k & 0x7FFFFFFFu) : ~k;
  return __uint_as_float(u);
}

__global__ void edge_max_kernel(const int* __restrict__ src, const int* __restrict__ dst,
                                const float* __restrict__ a_s, const float* __restrict__ a_d,
                                unsigned* __restrict__ mkey, int E) {
  int e = blockIdx.x * blockDim.x + threadIdx.x;
  if (e >= E) return;
  int s = src[e], d = dst[e];
#pragma unroll
  for (int h = 0; h < HEADS; ++h) {
    float x = a_s[s * 4 + h] + a_d[d * 4 + h];
    x = (x > 0.f) ? x : 0.2f * x;                  // leaky_relu(0.2)
    atomicMax(&mkey[d * 4 + h], f2key(x));
  }
}

__global__ void decode_m_kernel(const unsigned* __restrict__ mkey, float* __restrict__ m, int n) {
  int i = blockIdx.x * blockDim.x + threadIdx.x;
  if (i >= n) return;
  unsigned k = mkey[i];
  m[i] = (k == 0u) ? 0.f : key2f(k);               // no-edge segments -> 0 (matches ref)
}

__global__ void edge_den_kernel(const int* __restrict__ src, const int* __restrict__ dst,
                                const float* __restrict__ a_s, const float* __restrict__ a_d,
                                const float* __restrict__ m, float* __restrict__ den, int E) {
  int e = blockIdx.x * blockDim.x + threadIdx.x;
  if (e >= E) return;
  int s = src[e], d = dst[e];
#pragma unroll
  for (int h = 0; h < HEADS; ++h) {
    float x = a_s[s * 4 + h] + a_d[d * 4 + h];
    x = (x > 0.f) ? x : 0.2f * x;
    atomicAdd(&den[d * 4 + h], __expf(x - m[d * 4 + h]));
  }
}

// wave per edge: lane c handles channel c of each head
__global__ __launch_bounds__(256)
void edge_aggr_kernel(const int* __restrict__ src, const int* __restrict__ dst,
                      const float* __restrict__ a_s, const float* __restrict__ a_d,
                      const float* __restrict__ m, const float* __restrict__ den,
                      const float* __restrict__ hs, float* __restrict__ acc, int E) {
  int e = (int)(((long)blockIdx.x * blockDim.x + threadIdx.x) >> 5);
  int lane = threadIdx.x & 31;
  if (e >= E) return;
  int s = src[e], d = dst[e];
  float al[HEADS];
#pragma unroll
  for (int h = 0; h < HEADS; ++h) {
    float x = a_s[s * 4 + h] + a_d[d * 4 + h];
    x = (x > 0.f) ? x : 0.2f * x;
    al[h] = __expf(x - m[d * 4 + h]) / (den[d * 4 + h] + 1e-16f);
  }
  const float* hrow = hs + (size_t)s * HID;
  float* arow = acc + (size_t)d * HID;
#pragma unroll
  for (int h = 0; h < HEADS; ++h)
    atomicAdd(&arow[h * CH + lane], hrow[h * CH + lane] * al[h]);
}

// wave per node: h = LN(relu((acc + sum biases) * inv_cnt)) * g + b
__global__ __launch_bounds__(256)
void finalize_kernel(const float* __restrict__ acc,
                     const float* __restrict__ b0, const float* __restrict__ b1,
                     const float* __restrict__ b2, float inv_cnt,
                     const float* __restrict__ g, const float* __restrict__ beta,
                     float* __restrict__ out, int Nn) {
  int node = (int)(((long)blockIdx.x * blockDim.x + threadIdx.x) >> 5);
  int lane = threadIdx.x & 31;
  if (node >= Nn) return;
  const float* ar = acc + (size_t)node * HID;
  float r[4], s1 = 0.f, s2 = 0.f;
#pragma unroll
  for (int j = 0; j < 4; ++j) {
    int c = lane + 32 * j;
    float v = ar[c] + b0[c] + b1[c];
    if (b2) v += b2[c];
    v *= inv_cnt;
    v = fmaxf(v, 0.f);
    r[j] = v; s1 += v; s2 += v * v;
  }
#pragma unroll
  for (int off = 16; off > 0; off >>= 1) {
    s1 += __shfl_xor(s1, off, 32);
    s2 += __shfl_xor(s2, off, 32);
  }
  float mu  = s1 * (1.f / HID);
  float var = s2 * (1.f / HID) - mu * mu;
  float rs  = rsqrtf(var + 1e-5f);
  float* orow = out + (size_t)node * HID;
#pragma unroll
  for (int j = 0; j < 4; ++j) {
    int c = lane + 32 * j;
    orow[c] = (r[j] - mu) * rs * g[c] + beta[c];
  }
}

// wave per node: out[n] = sigmoid(dot(t[n], w2) + b2)
__global__ __launch_bounds__(256)
void head2_kernel(const float* __restrict__ t, const float* __restrict__ w2,
                  const float* __restrict__ b2, float* __restrict__ out, int Nn) {
  int node = (int)(((long)blockIdx.x * blockDim.x + threadIdx.x) >> 5);
  int lane = threadIdx.x & 31;
  if (node >= Nn) return;
  const float* tr = t + (size_t)node * HID;
  float s = 0.f;
#pragma unroll
  for (int j = 0; j < 4; ++j) { int c = lane + 32 * j; s += tr[c] * w2[c]; }
#pragma unroll
  for (int off = 16; off > 0; off >>= 1) s += __shfl_xor(s, off, 32);
  if (lane == 0) out[node] = 1.f / (1.f + __expf(-(s + b2[0])));
}

// ---------------------------------------------------------------------------
// host driver
// ---------------------------------------------------------------------------
extern "C" void kernel_launch(void* const* d_in, const int* in_sizes, int n_in,
                              void* d_out, int out_size, void* d_ws, size_t ws_size,
                              hipStream_t stream) {
  (void)n_in; (void)out_size; (void)ws_size;
  // node types: 0=document 1=statute 2=section 3=claim
  static const int  NNT[4]  = {10000, 5000, 50000, 30000};
  static const long ROFF[4] = {0, 10000, 15000, 65000};
  const int NTOT = 95000;
  const int E = in_sizes[4] / 2;

  const float* xin[4] = {(const float*)d_in[0], (const float*)d_in[1],
                         (const float*)d_in[2], (const float*)d_in[3]};
  const int* ei[3] = {(const int*)d_in[4], (const int*)d_in[5], (const int*)d_in[6]};

  // params flattened in JAX pytree (sorted-key) order starting at d_in[7]:
  //   conv: 2 layers x 10 sorted edge-types x {att_dst, att_src, bias, w_dst, w_src}
  //   emb : 4 sorted nts x {b, w} ; heads: 3 sorted x {l1.b,l1.w,l2.b,l2.w}
  //   ln  : 2 layers x 4 sorted nts x {b, g}
  auto PF = [&](int i) { return (const float*)d_in[i]; };
  auto conv_leaf = [&](int li, int t, int which) { return PF(7 + li * 50 + t * 5 + which); };
  static const int SPOS[4] = {1, 3, 2, 0};   // nt id -> sorted position (claim,doc,sec,stat)
  auto emb_b = [&](int nt) { return PF(107 + SPOS[nt] * 2); };
  auto emb_w = [&](int nt) { return PF(108 + SPOS[nt] * 2); };
  auto ln_b  = [&](int li, int nt) { return PF(127 + li * 8 + SPOS[nt] * 2); };
  auto ln_g  = [&](int li, int nt) { return PF(128 + li * 8 + SPOS[nt] * 2); };

  // workspace carve (~153 MB)
  uint8_t* wp = (uint8_t*)d_ws;
  auto carve = [&](size_t bytes) { uint8_t* p = wp; wp += (bytes + 255) & ~size_t(255); return p; };
  float*    x_cur = (float*)carve((size_t)NTOT * HID * 4);
  float*    accb  = (float*)carve((size_t)NTOT * HID * 4);
  float*    hs    = (float*)carve((size_t)50000 * HID * 4);
  float*    hd    = (float*)carve((size_t)50000 * HID * 4);
  float*    a_s   = (float*)carve((size_t)50000 * 4 * 4);
  float*    a_d   = (float*)carve((size_t)50000 * 4 * 4);
  unsigned* mkey  = (unsigned*)carve((size_t)50000 * 4 * 4);
  float*    mval  = (float*)carve((size_t)50000 * 4 * 4);
  float*    den   = (float*)carve((size_t)50000 * 4 * 4);
  __bf16*   wtA   = (__bf16*)carve((size_t)768 * 128 * 2);
  __bf16*   wtB   = (__bf16*)carve((size_t)128 * 128 * 2);

  auto zero = [&](void* p, size_t nfloats) {
    long n4 = (long)(nfloats / 4);
    zero_f4_kernel<<<dim3((unsigned)((n4 + 255) / 256)), dim3(256), 0, stream>>>((float4*)p, n4);
  };
  auto prep = [&](const float* W, __bf16* Wt, int K) {
    int n = K * HID;
    prep_wt_kernel<<<dim3((n + 255) / 256), dim3(256), 0, stream>>>(W, Wt, K, HID);
  };
  auto gemm = [&](const float* A, const __bf16* Wt, const float* bias, float* C,
                  int M, int K, int flags) {
    gemm_bf16_wmma_kernel<<<dim3((M + 127) / 128), dim3(256), 2 * BK * HID * 2, stream>>>(
        A, Wt, bias, C, M, K, flags);
  };

  // 1) embeddings: x[nt] = xin[nt] @ W_emb + b
  for (int nt = 0; nt < 4; ++nt) {
    prep(emb_w(nt), wtA, 768);
    gemm(xin[nt], wtA, emb_b(nt), x_cur + ROFF[nt] * HID, NNT[nt], 768, 0);
  }

  // sorted edge-type table: {src nt, dst nt, ei index, swap rows, self-loop}
  struct ETE { int s, d, eidx; bool swap, self; };
  static const ETE T[10] = {
    {3, 1, 0, false, false},  // claim cites statute
    {3, 0, 2, true,  false},  // claim rev_mentions document
    {3, 3, -1, false, true},  // claim self
    {0, 3, 2, false, false},  // document mentions claim
    {0, 0, -1, false, true},  // document self
    {2, 1, 1, true,  false},  // section rev_contains statute
    {2, 2, -1, false, true},  // section self
    {1, 2, 1, false, false},  // statute contains section
    {1, 3, 0, true,  false},  // statute rev_cites claim
    {1, 1, -1, false, true},  // statute self
  };

  for (int li = 0; li < 2; ++li) {
    zero(accb, (size_t)NTOT * HID);
    for (int t = 0; t < 10; ++t) {
      const ETE& et = T[t];
      const float* xs = x_cur + ROFF[et.s] * HID;
      const float* xd = x_cur + ROFF[et.d] * HID;
      const int Ns = NNT[et.s], Nd = NNT[et.d];
      if (et.self) {
        // single self-edge per dst -> softmax alpha == 1 -> acc += x @ w_src
        prep(conv_leaf(li, t, 4), wtA, HID);
        gemm(xs, wtA, nullptr, accb + ROFF[et.s] * HID, Ns, HID, 2);
        continue;
      }
      prep(conv_leaf(li, t, 4), wtA, HID);                         // w_src
      gemm(xs, wtA, nullptr, hs, Ns, HID, 0);
      prep(conv_leaf(li, t, 3), wtB, HID);                         // w_dst
      gemm(xd, wtB, nullptr, hd, Nd, HID, 0);
      att_reduce_kernel<<<dim3((Ns * 4 + 255) / 256), 256, 0, stream>>>(hs, conv_leaf(li, t, 1), a_s, Ns);
      att_reduce_kernel<<<dim3((Nd * 4 + 255) / 256), 256, 0, stream>>>(hd, conv_leaf(li, t, 0), a_d, Nd);
      zero(mkey, (size_t)Nd * 4);
      zero(den,  (size_t)Nd * 4);
      const int* esrc = ei[et.eidx] + (et.swap ? E : 0);
      const int* edst = ei[et.eidx] + (et.swap ? 0 : E);
      edge_max_kernel<<<dim3((E + 255) / 256), 256, 0, stream>>>(esrc, edst, a_s, a_d, mkey, E);
      decode_m_kernel<<<dim3((Nd * 4 + 255) / 256), 256, 0, stream>>>(mkey, mval, Nd * 4);
      edge_den_kernel<<<dim3((E + 255) / 256), 256, 0, stream>>>(esrc, edst, a_s, a_d, mval, den, E);
      edge_aggr_kernel<<<dim3((E + 7) / 8), 256, 0, stream>>>(esrc, edst, a_s, a_d, mval, den, hs,
                                                              accb + ROFF[et.d] * HID, E);
    }
    // incoming edge types per dst nt (sorted-t indices); -1 = absent
    const int BL[4][3] = {{1, 4, -1}, {0, 5, 9}, {6, 7, -1}, {2, 3, 8}};
    for (int nt = 0; nt < 4; ++nt) {
      if (li == 1 && nt != 3) continue;   // only claim needed after last layer
      const float* b0 = conv_leaf(li, BL[nt][0], 2);
      const float* b1 = conv_leaf(li, BL[nt][1], 2);
      const float* b2 = (BL[nt][2] >= 0) ? conv_leaf(li, BL[nt][2], 2) : nullptr;
      const float inv_cnt = (BL[nt][2] >= 0) ? (1.f / 3.f) : 0.5f;
      float* outp = (li == 1 && nt == 3) ? (float*)d_out : (x_cur + ROFF[nt] * HID);
      finalize_kernel<<<dim3((NNT[nt] + 7) / 8), 256, 0, stream>>>(
          accb + ROFF[nt] * HID, b0, b1, b2, inv_cnt, ln_g(li, nt), ln_b(li, nt), outp, NNT[nt]);
    }
  }

  // heads (output order: citation_validity, relevance_score, coherence_score;
  //        param order sorted: cv=115, coherence=119, relevance=123)
  const float* claimx = (const float*)d_out;
  const int NC = 30000;
  struct HD { int base; long off; };
  const HD H[3] = { {115, (long)NC * HID},
                    {123, (long)NC * HID + NC},
                    {119, (long)NC * HID + 2 * NC} };
  for (int i = 0; i < 3; ++i) {
    prep(PF(H[i].base + 1), wtB, HID);                              // l1.w
    gemm(claimx, wtB, PF(H[i].base + 0), hs, NC, HID, 1);           // relu fused
    head2_kernel<<<dim3((NC + 7) / 8), 256, 0, stream>>>(
        hs, PF(H[i].base + 3), PF(H[i].base + 2), (float*)d_out + H[i].off, NC);
  }
}
